// CrossAttention_5738076307519
// MI455X (gfx1250) — matrix operand order repaired and checked by reading, not compiled, not run on hardware
//
#include <hip/hip_runtime.h>
#include <hip/hip_bf16.h>

// ---------- types ----------
typedef __attribute__((ext_vector_type(16))) __bf16 v16bf;
typedef __attribute__((ext_vector_type(8)))  float  v8f;

union FragBF { v16bf v; unsigned int u[8]; };

__device__ __forceinline__ unsigned short f2bf(float f) {
    unsigned int u = __builtin_bit_cast(unsigned int, f);
    u += 0x7FFFu + ((u >> 16) & 1u);          // round-to-nearest-even
    return (unsigned short)(u >> 16);
}

__device__ __forceinline__ v8f zero8() {
    v8f z = {0.f, 0.f, 0.f, 0.f, 0.f, 0.f, 0.f, 0.f};
    return z;
}

__device__ __forceinline__ v8f wmma_bf16(v16bf a, v16bf b, v8f c) {
    return __builtin_amdgcn_wmma_f32_16x16x32_bf16(false, a, false, b, (short)0, c, false, false);
}

// A-matrix 16x32 bf16 fragment from row-major [m][k] storage (stride in halves).
// lane m = lane&15; kbase = (lane>=16)?16:0;
//   VGPR i : K = kbase + 2i, kbase + 2i + 1 (contiguous, matches load_frag_B)
__device__ __forceinline__ v16bf load_frag_A(const unsigned short* base, int stride, int lane) {
    const unsigned short* row = base + (lane & 15) * stride + ((lane & 16) ? 16 : 0);
    FragBF f;
#pragma unroll
    for (int i = 0; i < 8; ++i)
        f.u[i] = *(const unsigned int*)(row + 2 * i);
    return f.v;
}

// B-matrix 32x16 bf16 fragment from [n][k]-major storage (stride in halves).
// lane n = lane&15; here lanes 0-15 hold K=0..15, lanes 16-31 hold K=16..31,
// VGPR i holds K = kbase + 2i, kbase + 2i + 1 (contiguous pairs).
__device__ __forceinline__ v16bf load_frag_B(const unsigned short* base, int stride, int lane) {
    const unsigned short* row = base + (lane & 15) * stride + ((lane & 16) ? 16 : 0);
    FragBF f;
#pragma unroll
    for (int i = 0; i < 8; ++i)
        f.u[i] = *(const unsigned int*)(row + 2 * i);
    return f.v;
}

// ---------- constants ----------
#define HEADS 8
#define DH    64
#define QD    512
#define CD    768
#define NQ    4096
#define NBATCH 16
#define MCTX  77
#define MP80  80
#define MP96  96
#define HS    520   // LDS row stride (halves) for 512-wide tiles
#define AS    104   // LDS row stride (halves) for attn tiles (96 + pad)

// ---------- prep: transpose+convert Wq, Wo to bf16 [n][k] ----------
__global__ void prep_weights(const float* __restrict__ Wq, const float* __restrict__ Wo,
                             unsigned short* __restrict__ WqT, unsigned short* __restrict__ WoT) {
    int idx = blockIdx.x * blockDim.x + threadIdx.x;   // 512*512 threads
    int n = idx & 511, k = idx >> 9;
    WqT[n * QD + k] = f2bf(Wq[k * QD + n]);
    WoT[n * QD + k] = f2bf(Wo[k * QD + n]);
}

// ---------- prep: K/V projections to bf16, padded, fragment-friendly ----------
// kT: [B*H][80][64]  (rows m, contiguous d)  -> B operand of Q@K^T
// vT: [B*H][64][96]  (rows d, contiguous m)  -> B operand of attn@V
__global__ void prep_kv(const float* __restrict__ ctx, const float* __restrict__ Wk,
                        const float* __restrict__ Wv, unsigned short* __restrict__ kT,
                        unsigned short* __restrict__ vT) {
    int b = blockIdx.x >> 3, h = blockIdx.x & 7;
    int d = threadIdx.x & 63, msub = threadIdx.x >> 6;   // 64 d x 4 msub
    int col = h * DH + d;
    float ka[24], va[24];
#pragma unroll
    for (int j = 0; j < 24; ++j) { ka[j] = 0.f; va[j] = 0.f; }
    const float* cb = ctx + (size_t)b * MCTX * CD;
    for (int c = 0; c < CD; ++c) {
        float wk = Wk[c * QD + col];
        float wv = Wv[c * QD + col];
#pragma unroll
        for (int j = 0; j < 24; ++j) {
            int m = msub + 4 * j;
            if (m < MCTX) {
                float f = cb[(size_t)m * CD + c];
                ka[j] += f * wk;
                va[j] += f * wv;
            }
        }
    }
    size_t bh = (size_t)(b * HEADS + h);
#pragma unroll
    for (int j = 0; j < 24; ++j) {
        int m = msub + 4 * j;
        if (m < MP80) kT[(bh * MP80 + m) * DH + d] = f2bf(m < MCTX ? ka[j] : 0.f);
        if (m < MP96) vT[(bh * DH + d) * MP96 + m] = f2bf(m < MCTX ? va[j] : 0.f);
    }
}

// ---------- fused LN + Q-proj + attention + O-proj + residual ----------
__global__ __launch_bounds__(256)
void attn_main(const float* __restrict__ x, const float* __restrict__ gamma,
               const float* __restrict__ beta, const unsigned short* __restrict__ WqT,
               const unsigned short* __restrict__ WoT, const float* __restrict__ bo,
               const unsigned short* __restrict__ kT, const unsigned short* __restrict__ vT,
               float* __restrict__ out) {
    extern __shared__ unsigned short smem[];
    unsigned short* h_lds = smem;                     // [64][HS] bf16; reused for attn output
    unsigned short* q_lds = smem + 64 * HS;           // [64][HS] bf16
    unsigned short* a_lds = smem + 2 * 64 * HS;       // [8][64][AS] bf16 (first staged K, then attn)

    const int b   = blockIdx.y;
    const int q0  = blockIdx.x * 64;
    const int tid = threadIdx.x;
    const int wave = tid >> 5;
    const int lane = tid & 31;

    // ---- Async stage: copy this wave's K-head tile (80x64 bf16 = 10 KB) into its
    // attn LDS slice with ASYNCcnt-tracked copies; overlaps LN + Q-proj. ----
    {
        const unsigned short* kh = kT + (size_t)(b * HEADS + wave) * MP80 * DH;
        unsigned aoff = (unsigned)(wave * 64 * AS) * 2u;   // byte offset of slice in dynamic LDS
#pragma unroll
        for (int j = 0; j < 20; ++j) {                     // 640 x 16B chunks / 32 lanes
            unsigned off = (unsigned)(lane + 32 * j) * 16u;
            unsigned ldst = aoff + off;
            unsigned long long g = (unsigned long long)((const char*)kh + off);
            asm volatile("global_load_async_to_lds_b128 %0, %1, off"
                         :: "v"(ldst), "v"(g) : "memory");
        }
    }

    // ---- Step A: LayerNorm (fp32) -> h bf16 in LDS; 8 rows per wave ----
    for (int rr = 0; rr < 8; ++rr) {
        int row = wave * 8 + rr;
        const float* xr = x + ((size_t)b * NQ + q0 + row) * QD;
        float vals[16];
        float s = 0.f;
#pragma unroll
        for (int i = 0; i < 16; ++i) { vals[i] = xr[lane + 32 * i]; s += vals[i]; }
        for (int m = 16; m >= 1; m >>= 1) s += __shfl_xor(s, m, 32);
        float mu = s * (1.f / 512.f);
        float vs = 0.f;
#pragma unroll
        for (int i = 0; i < 16; ++i) { float d0 = vals[i] - mu; vs += d0 * d0; }
        for (int m = 16; m >= 1; m >>= 1) vs += __shfl_xor(vs, m, 32);
        float rstd = rsqrtf(vs * (1.f / 512.f) + 1e-5f);
        unsigned short* hr = h_lds + row * HS;
#pragma unroll
        for (int i = 0; i < 16; ++i) {
            int c0 = lane + 32 * i;
            hr[c0] = f2bf((vals[i] - mu) * rstd * gamma[c0] + beta[c0]);
        }
    }
    __syncthreads();

    // ---- Step B: Q = h @ Wq ; wave owns 4 row-tiles x 4 col-tiles (cols wave*64..) ----
    {
        const int cb = wave * 4;
        v8f acc[16];
#pragma unroll
        for (int t = 0; t < 16; ++t) acc[t] = zero8();
        for (int kc = 0; kc < 16; ++kc) {
            v16bf afr[4], bfr[4];
#pragma unroll
            for (int rt = 0; rt < 4; ++rt)
                afr[rt] = load_frag_A(h_lds + rt * 16 * HS + kc * 32, HS, lane);
#pragma unroll
            for (int ct = 0; ct < 4; ++ct)
                bfr[ct] = load_frag_B(WqT + (size_t)(cb + ct) * 16 * QD + kc * 32, QD, lane);
#pragma unroll
            for (int rt = 0; rt < 4; ++rt)
#pragma unroll
                for (int ct = 0; ct < 4; ++ct)
                    acc[rt * 4 + ct] = wmma_bf16(afr[rt], bfr[ct], acc[rt * 4 + ct]);
        }
#pragma unroll
        for (int rt = 0; rt < 4; ++rt)
#pragma unroll
            for (int ct = 0; ct < 4; ++ct) {
                int n_l = (cb + ct) * 16 + (lane & 15);
                int m0 = rt * 16 + ((lane & 16) ? 8 : 0);
#pragma unroll
                for (int r = 0; r < 8; ++r)
                    q_lds[(m0 + r) * HS + n_l] = f2bf(acc[rt * 4 + ct][r]);
            }
    }
    __syncthreads();

    // ---- Step C: per-head attention; wave = head ----
    {
        const int hh = wave;
        const unsigned short* vh = vT + (size_t)(b * HEADS + hh) * DH * MP96;  // [64][96]
        unsigned short* ah = a_lds + hh * 64 * AS;        // staged K now, attn later

        // wait for this wave's async K tile
        asm volatile("s_wait_asynccnt 0" ::: "memory");

        // scores = q_h @ k_h^T : 4 mt x 5 nt tiles, K = 64 ; K-frags from LDS ([80][64])
        v8f sc[20];
#pragma unroll
        for (int t = 0; t < 20; ++t) sc[t] = zero8();
        for (int kc = 0; kc < 2; ++kc) {
            v16bf afr[4];
#pragma unroll
            for (int mt = 0; mt < 4; ++mt)
                afr[mt] = load_frag_A(q_lds + mt * 16 * HS + hh * DH + kc * 32, HS, lane);
#pragma unroll
            for (int nt = 0; nt < 5; ++nt) {
                v16bf bfr = load_frag_B(ah + nt * 16 * DH + kc * 32, DH, lane);
#pragma unroll
                for (int mt = 0; mt < 4; ++mt)
                    sc[mt * 5 + nt] = wmma_bf16(afr[mt], bfr, sc[mt * 5 + nt]);
            }
        }

        // K tile consumed; zero attn K-padding cols 80..95 in this wave's slice
        for (int r = lane; r < 64; r += 32)
#pragma unroll
            for (int c2 = MP80; c2 < MP96; ++c2) ah[r * AS + c2] = 0;

        // softmax over context dim (mask cols >= 77); C layout: lane -> col, VGPR -> row
        const float scale = 0.125f;  // 1/sqrt(64)
        const int ncol = lane & 15;
#pragma unroll
        for (int mt = 0; mt < 4; ++mt) {
#pragma unroll
            for (int r = 0; r < 8; ++r) {
                float v5[5];
                float mx = -3.0e38f;
#pragma unroll
                for (int nt = 0; nt < 5; ++nt) {
                    float sv = sc[mt * 5 + nt][r] * scale;
                    if (nt * 16 + ncol >= MCTX) sv = -3.0e38f;
                    v5[nt] = sv;
                    mx = fmaxf(mx, sv);
                }
                for (int m2 = 8; m2 >= 1; m2 >>= 1) mx = fmaxf(mx, __shfl_xor(mx, m2, 32));
                float sum = 0.f;
#pragma unroll
                for (int nt = 0; nt < 5; ++nt) {
                    float e = __expf(v5[nt] - mx);
                    if (nt * 16 + ncol >= MCTX) e = 0.f;
                    v5[nt] = e;
                    sum += e;
                }
                for (int m2 = 8; m2 >= 1; m2 >>= 1) sum += __shfl_xor(sum, m2, 32);
                float inv = 1.f / sum;
                int mrow = mt * 16 + r + ((lane & 16) ? 8 : 0);
#pragma unroll
                for (int nt = 0; nt < 5; ++nt)
                    ah[mrow * AS + nt * 16 + ncol] = f2bf(v5[nt] * inv);
            }
        }

        // out_h = attn @ v_h : 4 mt x 4 nt tiles, K = 96
        v8f oc[16];
#pragma unroll
        for (int t = 0; t < 16; ++t) oc[t] = zero8();
        for (int kc = 0; kc < 3; ++kc) {
            v16bf bfr[4];
#pragma unroll
            for (int nt = 0; nt < 4; ++nt)
                bfr[nt] = load_frag_B(vh + nt * 16 * MP96 + kc * 32, MP96, lane);
#pragma unroll
            for (int mt = 0; mt < 4; ++mt) {
                v16bf afr = load_frag_A(ah + mt * 16 * AS + kc * 32, AS, lane);
#pragma unroll
                for (int nt = 0; nt < 4; ++nt)
                    oc[mt * 4 + nt] = wmma_bf16(afr, bfr[nt], oc[mt * 4 + nt]);
            }
        }
        __syncthreads();   // h_lds reads (step B) done everywhere before overwrite
#pragma unroll
        for (int mt = 0; mt < 4; ++mt)
#pragma unroll
            for (int nt = 0; nt < 4; ++nt) {
                int n_l = hh * DH + nt * 16 + (lane & 15);
                int m0 = mt * 16 + ((lane & 16) ? 8 : 0);
#pragma unroll
                for (int r = 0; r < 8; ++r)
                    h_lds[(m0 + r) * HS + n_l] = f2bf(oc[mt * 4 + nt][r]);
            }
    }
    __syncthreads();

    // ---- Step D: y = out_attn @ Wo + bo + x ; same 4x4 register blocking ----
    {
        const int cb = wave * 4;
        v8f acc[16];
#pragma unroll
        for (int t = 0; t < 16; ++t) acc[t] = zero8();
        for (int kc = 0; kc < 16; ++kc) {
            v16bf afr[4], bfr[4];
#pragma unroll
            for (int rt = 0; rt < 4; ++rt)
                afr[rt] = load_frag_A(h_lds + rt * 16 * HS + kc * 32, HS, lane);
#pragma unroll
            for (int ct = 0; ct < 4; ++ct)
                bfr[ct] = load_frag_B(WoT + (size_t)(cb + ct) * 16 * QD + kc * 32, QD, lane);
#pragma unroll
            for (int rt = 0; rt < 4; ++rt)
#pragma unroll
                for (int ct = 0; ct < 4; ++ct)
                    acc[rt * 4 + ct] = wmma_bf16(afr[rt], bfr[ct], acc[rt * 4 + ct]);
        }
#pragma unroll
        for (int rt = 0; rt < 4; ++rt)
#pragma unroll
            for (int ct = 0; ct < 4; ++ct) {
                int n_l = (cb + ct) * 16 + (lane & 15);
                float bias = bo[n_l];
                int m0 = rt * 16 + ((lane & 16) ? 8 : 0);
#pragma unroll
                for (int r = 0; r < 8; ++r) {
                    size_t gi = ((size_t)b * NQ + q0 + m0 + r) * QD + n_l;
                    out[gi] = x[gi] + acc[rt * 4 + ct][r] + bias;
                }
            }
    }
}

// ---------- launch ----------
extern "C" void kernel_launch(void* const* d_in, const int* in_sizes, int n_in,
                              void* d_out, int out_size, void* d_ws, size_t ws_size,
                              hipStream_t stream) {
    (void)in_sizes; (void)n_in; (void)out_size; (void)ws_size;
    const float* x     = (const float*)d_in[0];
    const float* ctx   = (const float*)d_in[1];
    const float* gamma = (const float*)d_in[2];
    const float* beta  = (const float*)d_in[3];
    const float* Wq    = (const float*)d_in[4];
    const float* Wk    = (const float*)d_in[5];
    const float* Wv    = (const float*)d_in[6];
    const float* Wo    = (const float*)d_in[7];
    const float* bo    = (const float*)d_in[8];
    float* out = (float*)d_out;

    unsigned short* WqT = (unsigned short*)d_ws;
    unsigned short* WoT = WqT + (size_t)QD * QD;                  // 512*512
    unsigned short* kT  = WoT + (size_t)QD * QD;
    unsigned short* vT  = kT + (size_t)NBATCH * HEADS * MP80 * DH;
    // total ws usage: 2*512*512 + 16*8*(80*64 + 64*96) halves ~= 3.93 MB

    prep_weights<<<(QD * QD) / 256, 256, 0, stream>>>(Wq, Wo, WqT, WoT);
    prep_kv<<<NBATCH * HEADS, 256, 0, stream>>>(ctx, Wk, Wv, kT, vT);

    size_t smem = (size_t)(2 * 64 * HS + HEADS * 64 * AS) * sizeof(unsigned short); // ~234 KB
    attn_main<<<dim3(NQ / 64, NBATCH), 256, smem, stream>>>(x, gamma, beta, WqT, WoT, bo, kT, vT, out);
}
